// LinAngularXCA_CA_16827681865776
// MI455X (gfx1250) — compile-verified, matrix operand order checked by reading, not run on hardware
//
#include <hip/hip_runtime.h>
#include <hip/hip_bf16.h>
#include <float.h>
#include <math.h>

// ---------------------------------------------------------------------------
// Problem constants (from reference)
// ---------------------------------------------------------------------------
#define B_      64
#define C_      32
#define H_      80
#define W_      80
#define L_      6400
#define HEADS_  8
#define HD_     4
#define INV_PI_ 0.31830988618379067f

// ---------------------------------------------------------------------------
// CDNA5 WMMA helpers (wave32; 16x16x32 f16 -> f32), CDNA5 ISA.
// Fragments are staged in LDS in per-lane layout: frag[lane][16 halfs], so a
// fragment load is one contiguous 32B vector read (2 x ds_load_b128).
//   A (16x32): lane m (0-15) = row m, halves = K {0..7,16..23}; lanes 16-31 =
//              same rows, K {8..15,24..31}.  B (32x16): same with lane = col n.
//   D (16x16 f32): lanes 0-15 -> N=lane, VGPR r -> M=r; lanes 16-31 -> M=8+r.
// ---------------------------------------------------------------------------
typedef __attribute__((ext_vector_type(16))) _Float16 v16h;
typedef __attribute__((ext_vector_type(8)))  _Float16 v8h;
typedef __attribute__((ext_vector_type(8)))  float    v8f;

// write one element into a fragment-major tile; rc = row (A) or column (B)
__device__ __forceinline__ void stage_frag(_Float16* frag, int rc, int k,
                                           _Float16 val) {
  int lane = rc + (((k & 15) >= 8) ? 16 : 0);
  int half = (k & 7) + ((k >= 16) ? 8 : 0);
  frag[lane * 16 + half] = val;
}

__device__ __forceinline__ v16h load_frag(const _Float16* frag, int lane) {
  return *(const v16h*)(frag + lane * 16);
}

__device__ __forceinline__ void store_d(float* base, int ld, int lane, v8f d) {
  int n  = lane & 15;
  int m0 = (lane & 16) ? 8 : 0;
#pragma unroll
  for (int r = 0; r < 8; ++r) base[(m0 + r) * ld + n] = d[r];
}

__device__ __forceinline__ v8f wmma16(v16h a, v16h b, v8f c) {
  return __builtin_amdgcn_wmma_f32_16x16x32_f16(false, a, false, b,
                                                (short)0, c, false, false);
}

__device__ __forceinline__ void atomicMaxF(float* addr, float val) {
  unsigned int* ua  = (unsigned int*)addr;
  unsigned int  cur = *ua;
  while (__uint_as_float(cur) < val) {
    unsigned int prev = atomicCAS(ua, cur, __float_as_uint(val));
    if (prev == cur) break;
    cur = prev;
  }
}

// ---------------------------------------------------------------------------
// fill kernel (accumulator init; d_ws is not re-poisoned between replays)
// ---------------------------------------------------------------------------
__global__ void k_fill(float* p, float v, int n) {
  int i = blockIdx.x * 256 + threadIdx.x;
  if (i < n) p[i] = v;
}

// ---------------------------------------------------------------------------
// Kernel 1: LA branch. qkv = x1 @ W^T via WMMA (K=32 == one wmma), per-token
// q/k normalization over HD=4, store normalized q and raw v (f16), and
// reduce attn1[b,h,d,e] = sum_l kn_d * v_e without LDS atomics.
// ---------------------------------------------------------------------------
__global__ __launch_bounds__(256) void k_qkv_la(
    const float* __restrict__ rgb, const float* __restrict__ qkv_w,
    _Float16* __restrict__ q1n, _Float16* __restrict__ v1,
    float* __restrict__ attn1) {
  __shared__ alignas(32) _Float16 xfrag[4][32 * 16];   // 4 row slabs
  __shared__ alignas(32) _Float16 bfrag[6][32 * 16];   // 6 N tiles
  __shared__ float qs[64][100];

  const int b    = blockIdx.y;
  const int l0   = blockIdx.x * 64;
  const int tid  = threadIdx.x;
  const int lane = tid & 31;
  const int wave = tid >> 5;

  // stage x tile (transpose rgb[b,c,l] on the fly) straight into frag layout
  for (int idx = tid; idx < 64 * 32; idx += 256) {
    int r = idx & 63, c = idx >> 6;
    const float* src = &rgb[((size_t)b * C_ + c) * L_ + l0 + r];
    if (l0 + 64 < L_) __builtin_prefetch(src + 64, 0, 1);
    stage_frag(&xfrag[r >> 4][0], r & 15, c, (_Float16)(*src));
  }
  // stage W^T: element (k=c_in, n=c_out) = qkv_w[n][k]
  for (int idx = tid; idx < 32 * 96; idx += 256) {
    int k = idx & 31, n = idx >> 5;
    stage_frag(&bfrag[n >> 4][0], n & 15, k, (_Float16)qkv_w[n * 32 + k]);
  }
  __syncthreads();

  {  // 24 WMMA tiles: wave -> (slab of 16 rows, 3 of 6 N-tiles)
    const int slab = wave >> 1;
    const int nb   = (wave & 1) * 3;
    v16h a = load_frag(xfrag[slab], lane);
#pragma unroll
    for (int j = 0; j < 3; ++j) {
      v16h bf = load_frag(bfrag[nb + j], lane);
      v8f  d  = {};
      d = wmma16(a, bf, d);
      store_d(&qs[slab * 16][(nb + j) * 16], 100, lane, d);
    }
  }
  __syncthreads();

  // phase 1: per-row normalization; write q1n/v1 (16B packets); kn -> qs
  if (tid < 64) {
    const int    r    = tid;
    const size_t rowo = ((size_t)b * L_ + l0 + r) * 32;
    alignas(16) _Float16 qbuf[32], vbuf[32];
#pragma unroll
    for (int h = 0; h < HEADS_; ++h) {
      float4 q4 = *(const float4*)&qs[r][h * 4];
      float4 k4 = *(const float4*)&qs[r][32 + h * 4];
      float4 v4 = *(const float4*)&qs[r][64 + h * 4];
      float  rq = rsqrtf(q4.x * q4.x + q4.y * q4.y + q4.z * q4.z + q4.w * q4.w);
      float  rk = rsqrtf(k4.x * k4.x + k4.y * k4.y + k4.z * k4.z + k4.w * k4.w);
      qbuf[h * 4 + 0] = (_Float16)(q4.x * rq);
      qbuf[h * 4 + 1] = (_Float16)(q4.y * rq);
      qbuf[h * 4 + 2] = (_Float16)(q4.z * rq);
      qbuf[h * 4 + 3] = (_Float16)(q4.w * rq);
      vbuf[h * 4 + 0] = (_Float16)v4.x;
      vbuf[h * 4 + 1] = (_Float16)v4.y;
      vbuf[h * 4 + 2] = (_Float16)v4.z;
      vbuf[h * 4 + 3] = (_Float16)v4.w;
      qs[r][32 + h * 4 + 0] = k4.x * rk;   // in-place normalized k
      qs[r][32 + h * 4 + 1] = k4.y * rk;
      qs[r][32 + h * 4 + 2] = k4.z * rk;
      qs[r][32 + h * 4 + 3] = k4.w * rk;
    }
#pragma unroll
    for (int i = 0; i < 4; ++i) {
      *(v8h*)(q1n + rowo + i * 8) = *(const v8h*)&qbuf[i * 8];
      *(v8h*)(v1  + rowo + i * 8) = *(const v8h*)&vbuf[i * 8];
    }
  }
  __syncthreads();

  // phase 2: attn1[h,d,e] = sum_r kn[r,h,d] * v[r,h,e] (no LDS atomics)
  if (tid < 128) {
    int   h = tid >> 4, d = (tid >> 2) & 3, e = tid & 3;
    float s = 0.f;
#pragma unroll 4
    for (int r = 0; r < 64; ++r)
      s += qs[r][32 + h * 4 + d] * qs[r][64 + h * 4 + e];
    atomicAdd(&attn1[(size_t)b * 128 + tid], s);
  }
}

// ---------------------------------------------------------------------------
// Kernel 2: XCA branch qkv GEMM + reductions over L:
//   Sqq[b,h,d]=sum q^2, Skk[b,h,e]=sum k^2, Sqk[b,h,d,e]=sum q_d*k_e.
// ---------------------------------------------------------------------------
__global__ __launch_bounds__(256) void k_qkv_xca(
    const float* __restrict__ freq, const float* __restrict__ qkv_w,
    _Float16* __restrict__ v2, float* __restrict__ Sqk,
    float* __restrict__ Sqq, float* __restrict__ Skk) {
  __shared__ alignas(32) _Float16 xfrag[4][32 * 16];
  __shared__ alignas(32) _Float16 bfrag[6][32 * 16];
  __shared__ float qs[64][100];

  const int b    = blockIdx.y;
  const int l0   = blockIdx.x * 64;
  const int tid  = threadIdx.x;
  const int lane = tid & 31;
  const int wave = tid >> 5;

  for (int idx = tid; idx < 64 * 32; idx += 256) {
    int r = idx & 63, c = idx >> 6;
    const float* src = &freq[((size_t)b * C_ + c) * L_ + l0 + r];
    if (l0 + 64 < L_) __builtin_prefetch(src + 64, 0, 1);
    stage_frag(&xfrag[r >> 4][0], r & 15, c, (_Float16)(*src));
  }
  for (int idx = tid; idx < 32 * 96; idx += 256) {
    int k = idx & 31, n = idx >> 5;
    stage_frag(&bfrag[n >> 4][0], n & 15, k, (_Float16)qkv_w[n * 32 + k]);
  }
  __syncthreads();

  {
    const int slab = wave >> 1;
    const int nb   = (wave & 1) * 3;
    v16h a = load_frag(xfrag[slab], lane);
#pragma unroll
    for (int j = 0; j < 3; ++j) {
      v16h bf = load_frag(bfrag[nb + j], lane);
      v8f  d  = {};
      d = wmma16(a, bf, d);
      store_d(&qs[slab * 16][(nb + j) * 16], 100, lane, d);
    }
  }
  __syncthreads();

  // phase 1: store v (16B packets)
  if (tid < 64) {
    const int    r    = tid;
    const size_t rowo = ((size_t)b * L_ + l0 + r) * 32;
    alignas(16) _Float16 vbuf[32];
#pragma unroll
    for (int c = 0; c < 32; ++c) vbuf[c] = (_Float16)qs[r][64 + c];
#pragma unroll
    for (int i = 0; i < 4; ++i)
      *(v8h*)(v2 + rowo + i * 8) = *(const v8h*)&vbuf[i * 8];
  }
  __syncthreads();

  // phase 2a: Sqk[h,d,e] = sum_r q[r,h,d]*k[r,h,e]
  if (tid < 128) {
    int   h = tid >> 4, d = (tid >> 2) & 3, e = tid & 3;
    float s = 0.f;
#pragma unroll 4
    for (int r = 0; r < 64; ++r)
      s += qs[r][h * 4 + d] * qs[r][32 + h * 4 + e];
    atomicAdd(&Sqk[(size_t)b * 128 + tid], s);
  } else if (tid < 192) {
    // phase 2b: Sqq (tid 128..159) and Skk (tid 160..191)
    int   t   = tid - 128;
    int   c   = t & 31;
    int   off = (t < 32) ? 0 : 32;
    float s   = 0.f;
#pragma unroll 4
    for (int r = 0; r < 64; ++r) {
      float x = qs[r][off + c];
      s += x * x;
    }
    atomicAdd(((t < 32) ? Sqq : Skk) + (size_t)b * 32 + c, s);
  }
}

// ---------------------------------------------------------------------------
// Kernel 3: XCA 4x4 attention: normalize by L2 norms, *temp, softmax over e.
// ---------------------------------------------------------------------------
__global__ void k_attn2(const float* __restrict__ Sqk,
                        const float* __restrict__ Sqq,
                        const float* __restrict__ Skk,
                        const float* __restrict__ temp,
                        float* __restrict__ attn2) {
  __shared__ float s[16];
  int bh = blockIdx.x;
  int h  = bh & 7;
  int t  = threadIdx.x;
  int d  = t >> 2, e = t & 3;
  float nq  = fmaxf(sqrtf(Sqq[bh * 4 + d]), 1e-12f);
  float nk  = fmaxf(sqrtf(Skk[bh * 4 + e]), 1e-12f);
  float val = Sqk[bh * 16 + t] / (nq * nk) * temp[h];
  s[t] = val;
  __syncthreads();
  float m  = fmaxf(fmaxf(s[d * 4 + 0], s[d * 4 + 1]),
                   fmaxf(s[d * 4 + 2], s[d * 4 + 3]));
  float ex = __expf(val - m);
  __syncthreads();
  s[t] = ex;
  __syncthreads();
  float sum = s[d * 4 + 0] + s[d * 4 + 1] + s[d * 4 + 2] + s[d * 4 + 3];
  attn2[bh * 16 + t] = ex / sum;
}

// ---------------------------------------------------------------------------
// Kernel 4: LA output: o = 0.5 v + q@attn/pi, L2-normalize per head, add
// 9-tap depthwise conv over L (halo staged in LDS), WMMA projection, write
// f1 (B,C,L) with fused channel sum/max pooling.
// ---------------------------------------------------------------------------
__global__ __launch_bounds__(256) void k_la_out(
    const _Float16* __restrict__ q1n, const _Float16* __restrict__ v1,
    const float* __restrict__ attn1, const float* __restrict__ dconv_w,
    const float* __restrict__ proj_w, const float* __restrict__ proj_b,
    float* __restrict__ f1, float* __restrict__ sum1,
    float* __restrict__ max1) {
  __shared__ float att[HEADS_ * 16];
  __shared__ float dcw[HEADS_ * 9];
  __shared__ alignas(32) _Float16 pbfrag[2][32 * 16];
  __shared__ alignas(32) _Float16 ofrag[4][32 * 16];
  __shared__ alignas(32) _Float16 vh[72][32];       // v halo rows l0-4..l0+67
  __shared__ float res[64][36];
  __shared__ float redsum[32][8], redmax[32][8];

  const int b    = blockIdx.y;
  const int l0   = blockIdx.x * 64;
  const int tid  = threadIdx.x;
  const int lane = tid & 31;
  const int wave = tid >> 5;

  if (tid < 128) att[tid] = attn1[(size_t)b * 128 + tid];
  if (tid < 72)  dcw[tid] = dconv_w[tid];
  for (int idx = tid; idx < 1024; idx += 256) {
    int k = idx & 31, n = idx >> 5;
    stage_frag(&pbfrag[n >> 4][0], n & 15, k, (_Float16)proj_w[n * 32 + k]);
  }
  // stage v halo: 72 rows x 32 halfs, 16B packets
  for (int idx = tid; idx < 72 * 4; idx += 256) {
    int r = idx >> 2, seg = idx & 3;
    int lp = l0 + r - 4;
    v8h val;
    if (lp >= 0 && lp < L_) {
      val = *(const v8h*)(v1 + ((size_t)b * L_ + lp) * 32 + seg * 8);
    } else {
#pragma unroll
      for (int i = 0; i < 8; ++i) val[i] = (_Float16)0.f;
    }
    *(v8h*)&vh[r][seg * 8] = val;
  }
  __syncthreads();

  if (tid < 64) {
    const int    r    = tid;
    const size_t rowo = ((size_t)b * L_ + l0 + r) * 32;
    alignas(16) _Float16 qrow[32];
#pragma unroll
    for (int i = 0; i < 4; ++i)
      *(v8h*)&qrow[i * 8] = *(const v8h*)(q1n + rowo + i * 8);
#pragma unroll
    for (int h = 0; h < HEADS_; ++h) {
      float qv[4], vv[4], o[4], ss = 0.f;
#pragma unroll
      for (int d = 0; d < 4; ++d) {
        qv[d] = (float)qrow[h * 4 + d];
        vv[d] = (float)vh[r + 4][h * 4 + d];
      }
#pragma unroll
      for (int e = 0; e < 4; ++e) {
        float acc = 0.5f * vv[e];
#pragma unroll
        for (int d = 0; d < 4; ++d)
          acc += INV_PI_ * qv[d] * att[h * 16 + d * 4 + e];
        o[e] = acc;
        ss += acc * acc;
      }
      float rn = rsqrtf(ss);
#pragma unroll
      for (int e = 0; e < 4; ++e) {
        float dc = 0.f;
#pragma unroll
        for (int t9 = 0; t9 < 9; ++t9)
          dc += dcw[h * 9 + t9] * (float)vh[r + t9][h * 4 + e];
        stage_frag(&ofrag[r >> 4][0], r & 15, h * 4 + e,
                   (_Float16)(o[e] * rn + dc));
      }
    }
  }
  __syncthreads();

  {  // projection: 8 waves, one 16x16 tile each (4 slabs x 2 N-tiles)
    const int slab = wave >> 1;
    const int nt   = wave & 1;
    v16h a  = load_frag(ofrag[slab], lane);
    v16h bf = load_frag(pbfrag[nt], lane);
    v8f  d  = {};
    d = wmma16(a, bf, d);
    store_d(&res[slab * 16][nt * 16], 36, lane, d);
  }
  __syncthreads();

  // write f1 (float4) + fused per-channel sum/max partials
  {
    int   c  = tid >> 3;          // channel
    int   g  = tid & 7;           // row group
    int   r0 = g * 8;
    float bb = proj_b[c];
    float s = 0.f, mx = -FLT_MAX;
    float vals[8];
#pragma unroll
    for (int j = 0; j < 8; ++j) {
      float v = res[r0 + j][c] + bb;
      vals[j] = v;
      s += v;
      mx = fmaxf(mx, v);
    }
    float* dst = &f1[((size_t)b * 32 + c) * L_ + l0 + r0];
#pragma unroll
    for (int j = 0; j < 2; ++j)
      *(float4*)(dst + j * 4) = make_float4(vals[j * 4 + 0], vals[j * 4 + 1],
                                            vals[j * 4 + 2], vals[j * 4 + 3]);
    redsum[c][g] = s;
    redmax[c][g] = mx;
  }
  __syncthreads();
  if (tid < 32) {
    float s = 0.f, mx = -FLT_MAX;
#pragma unroll
    for (int g = 0; g < 8; ++g) {
      s += redsum[tid][g];
      mx = fmaxf(mx, redmax[tid][g]);
    }
    atomicAdd(&sum1[b * 32 + tid], s);
    atomicMaxF(&max1[b * 32 + tid], mx);
  }
}

// ---------------------------------------------------------------------------
// Kernel 5: XCA output: out[l, h*4+d] = sum_e attn2[h,d,e]*v2[l,h*4+e],
// then WMMA projection; writes f2 (B,C,L) + channel pooling.
// ---------------------------------------------------------------------------
__global__ __launch_bounds__(256) void k_xca_out(
    const _Float16* __restrict__ v2, const float* __restrict__ attn2,
    const float* __restrict__ proj_w, const float* __restrict__ proj_b,
    float* __restrict__ f2, float* __restrict__ sum2,
    float* __restrict__ max2) {
  __shared__ float att[HEADS_ * 16];
  __shared__ alignas(32) _Float16 pbfrag[2][32 * 16];
  __shared__ alignas(32) _Float16 ofrag[4][32 * 16];
  __shared__ float res[64][36];
  __shared__ float redsum[32][8], redmax[32][8];

  const int b    = blockIdx.y;
  const int l0   = blockIdx.x * 64;
  const int tid  = threadIdx.x;
  const int lane = tid & 31;
  const int wave = tid >> 5;

  if (tid < 128) att[tid] = attn2[(size_t)b * 128 + tid];
  for (int idx = tid; idx < 1024; idx += 256) {
    int k = idx & 31, n = idx >> 5;
    stage_frag(&pbfrag[n >> 4][0], n & 15, k, (_Float16)proj_w[n * 32 + k]);
  }
  __syncthreads();

  if (tid < 64) {
    const int    r    = tid;
    const size_t rowo = ((size_t)b * L_ + l0 + r) * 32;
    alignas(16) _Float16 vrow[32];
#pragma unroll
    for (int i = 0; i < 4; ++i)
      *(v8h*)&vrow[i * 8] = *(const v8h*)(v2 + rowo + i * 8);
#pragma unroll
    for (int h = 0; h < HEADS_; ++h) {
      float vv[4];
#pragma unroll
      for (int e = 0; e < 4; ++e) vv[e] = (float)vrow[h * 4 + e];
#pragma unroll
      for (int d = 0; d < 4; ++d) {
        float acc = 0.f;
#pragma unroll
        for (int e = 0; e < 4; ++e) acc += att[h * 16 + d * 4 + e] * vv[e];
        stage_frag(&ofrag[r >> 4][0], r & 15, h * 4 + d, (_Float16)acc);
      }
    }
  }
  __syncthreads();

  {
    const int slab = wave >> 1;
    const int nt   = wave & 1;
    v16h a  = load_frag(ofrag[slab], lane);
    v16h bf = load_frag(pbfrag[nt], lane);
    v8f  d  = {};
    d = wmma16(a, bf, d);
    store_d(&res[slab * 16][nt * 16], 36, lane, d);
  }
  __syncthreads();

  {
    int   c  = tid >> 3;
    int   g  = tid & 7;
    int   r0 = g * 8;
    float bb = proj_b[c];
    float s = 0.f, mx = -FLT_MAX;
    float vals[8];
#pragma unroll
    for (int j = 0; j < 8; ++j) {
      float v = res[r0 + j][c] + bb;
      vals[j] = v;
      s += v;
      mx = fmaxf(mx, v);
    }
    float* dst = &f2[((size_t)b * 32 + c) * L_ + l0 + r0];
#pragma unroll
    for (int j = 0; j < 2; ++j)
      *(float4*)(dst + j * 4) = make_float4(vals[j * 4 + 0], vals[j * 4 + 1],
                                            vals[j * 4 + 2], vals[j * 4 + 3]);
    redsum[c][g] = s;
    redmax[c][g] = mx;
  }
  __syncthreads();
  if (tid < 32) {
    float s = 0.f, mx = -FLT_MAX;
#pragma unroll
    for (int g = 0; g < 8; ++g) {
      s += redsum[tid][g];
      mx = fmaxf(mx, redmax[tid][g]);
    }
    atomicAdd(&sum2[b * 32 + tid], s);
    atomicMaxF(&max2[b * 32 + tid], mx);
  }
}

// ---------------------------------------------------------------------------
// Kernel 6: channel MLPs (32->16->32, avg+max branches) and cross-softmax
// weights w1[b,c,d] = softmax_d(a1[c]*a2[d]), w2[b,c,d] = softmax_d(a1[d]*a2[c])
// ---------------------------------------------------------------------------
__global__ __launch_bounds__(32) void k_chan_mlp(
    const float* sum1, const float* max1, const float* sum2, const float* max2,
    const float* avg1_w, const float* avg1_b, const float* max1_w,
    const float* max1_b, const float* avg11_w, const float* avg11_b,
    const float* max11_w, const float* max11_b, const float* avg2_w,
    const float* avg2_b, const float* max2_w, const float* max2_b,
    const float* avg22_w, const float* avg22_b, const float* max22_w,
    const float* max22_b, float* w1, float* w2) {
  __shared__ float avg[32], mxv[32], ha[16], hm[16], a1s[32], a2s[32];
  int b = blockIdx.x, c = threadIdx.x;

  avg[c] = sum1[b * 32 + c] * (1.0f / L_);
  mxv[c] = max1[b * 32 + c];
  __syncthreads();
  if (c < 16) {
    float s = avg1_b[c], t = max1_b[c];
    for (int k = 0; k < 32; ++k) {
      s += avg[k] * avg1_w[c * 32 + k];
      t += mxv[k] * max1_w[c * 32 + k];
    }
    ha[c] = fmaxf(s, 0.f);
    hm[c] = fmaxf(t, 0.f);
  }
  __syncthreads();
  {
    float a = avg11_b[c], m = max11_b[c];
    for (int j = 0; j < 16; ++j) {
      a += ha[j] * avg11_w[c * 16 + j];
      m += hm[j] * max11_w[c * 16 + j];
    }
    a1s[c] = a + m;
  }
  __syncthreads();

  avg[c] = sum2[b * 32 + c] * (1.0f / L_);
  mxv[c] = max2[b * 32 + c];
  __syncthreads();
  if (c < 16) {
    float s = avg2_b[c], t = max2_b[c];
    for (int k = 0; k < 32; ++k) {
      s += avg[k] * avg2_w[c * 32 + k];
      t += mxv[k] * max2_w[c * 32 + k];
    }
    ha[c] = fmaxf(s, 0.f);
    hm[c] = fmaxf(t, 0.f);
  }
  __syncthreads();
  {
    float a = avg22_b[c], m = max22_b[c];
    for (int j = 0; j < 16; ++j) {
      a += ha[j] * avg22_w[c * 16 + j];
      m += hm[j] * max22_w[c * 16 + j];
    }
    a2s[c] = a + m;
  }
  __syncthreads();

  {  // w1 row c
    float ac = a1s[c], mval = -FLT_MAX;
    for (int d = 0; d < 32; ++d) mval = fmaxf(mval, ac * a2s[d]);
    float ssum = 0.f;
    for (int d = 0; d < 32; ++d) ssum += __expf(ac * a2s[d] - mval);
    float inv = 1.0f / ssum;
    for (int d = 0; d < 32; ++d)
      w1[(size_t)b * 1024 + c * 32 + d] = __expf(ac * a2s[d] - mval) * inv;
  }
  {  // w2 row c
    float ac = a2s[c], mval = -FLT_MAX;
    for (int d = 0; d < 32; ++d) mval = fmaxf(mval, a1s[d] * ac);
    float ssum = 0.f;
    for (int d = 0; d < 32; ++d) ssum += __expf(a1s[d] * ac - mval);
    float inv = 1.0f / ssum;
    for (int d = 0; d < 32; ++d)
      w2[(size_t)b * 1024 + c * 32 + d] = __expf(a1s[d] * ac - mval) * inv;
  }
}

// ---------------------------------------------------------------------------
// Kernel 7 (x2): CAFM mixing a_f = softmax(cross) @ f via WMMA (M=32 chans,
// K=32, N=128 L-cols) fused with spatial pooling over channels (mean+max).
// a_f is never materialized in HBM -- only pooled (B,L) maps.
// ---------------------------------------------------------------------------
__global__ __launch_bounds__(256) void k_gate_pool(
    const float* __restrict__ w, const float* __restrict__ f,
    float* __restrict__ pmean, float* __restrict__ pmax) {
  __shared__ alignas(32) _Float16 wfrag[2][32 * 16];
  __shared__ alignas(32) _Float16 ffrag[8][32 * 16];
  __shared__ float res[32][136];

  const int b    = blockIdx.y;
  const int l0   = blockIdx.x * 128;
  const int tid  = threadIdx.x;
  const int lane = tid & 31;
  const int wave = tid >> 5;

  for (int idx = tid; idx < 1024; idx += 256) {
    int m = idx >> 5, k = idx & 31;
    stage_frag(&wfrag[m >> 4][0], m & 15, k,
               (_Float16)w[(size_t)b * 1024 + m * 32 + k]);
  }
  // stage f tile (K=32 channels x N=128 cols) via float4 reads
  for (int idx = tid; idx < 1024; idx += 256) {
    int    k  = idx >> 5;
    int    n0 = (idx & 31) * 4;
    float4 v4 = *(const float4*)&f[((size_t)b * 32 + k) * L_ + l0 + n0];
    stage_frag(&ffrag[n0 >> 4][0], (n0 + 0) & 15, k, (_Float16)v4.x);
    stage_frag(&ffrag[n0 >> 4][0], (n0 + 1) & 15, k, (_Float16)v4.y);
    stage_frag(&ffrag[n0 >> 4][0], (n0 + 2) & 15, k, (_Float16)v4.z);
    stage_frag(&ffrag[n0 >> 4][0], (n0 + 3) & 15, k, (_Float16)v4.w);
  }
  __syncthreads();

  {  // 16 tiles (2 M x 8 N) over 8 waves, 2 each
    const int mt  = wave & 1;
    const int nt0 = (wave >> 1) * 2;
    v16h a = load_frag(wfrag[mt], lane);
#pragma unroll
    for (int j = 0; j < 2; ++j) {
      v16h bf = load_frag(ffrag[nt0 + j], lane);
      v8f  d  = {};
      d = wmma16(a, bf, d);
      store_d(&res[mt * 16][(nt0 + j) * 16], 136, lane, d);
    }
  }
  __syncthreads();

  if (tid < 128) {
    int   n = tid;
    float s = 0.f, mx = -FLT_MAX;
#pragma unroll
    for (int c = 0; c < 32; ++c) {
      float v = res[c][n];
      s += v;
      mx = fmaxf(mx, v);
    }
    pmean[(size_t)b * L_ + l0 + n] = s * (1.0f / 32.0f);
    pmax [(size_t)b * L_ + l0 + n] = mx;
  }
}

// ---------------------------------------------------------------------------
// Kernels 8/9: spatial-gate 3x3 convs (2ch->1ch + relu, then 1ch->1ch)
// ---------------------------------------------------------------------------
__global__ void k_gate_conv1(const float* __restrict__ pmean,
                             const float* __restrict__ pmax,
                             const float* __restrict__ w,
                             const float* __restrict__ bias,
                             float* __restrict__ t1) {
  int i = blockIdx.x * 256 + threadIdx.x;
  if (i >= B_ * L_) return;
  int b = i / L_, p = i % L_;
  int y = p / W_, x = p % W_;
  float acc = bias[0];
#pragma unroll
  for (int dy = 0; dy < 3; ++dy)
#pragma unroll
    for (int dx = 0; dx < 3; ++dx) {
      int yy = y + dy - 1, xx = x + dx - 1;
      if (yy < 0 || yy >= H_ || xx < 0 || xx >= W_) continue;
      size_t q = (size_t)b * L_ + yy * W_ + xx;
      acc += w[dy * 3 + dx] * pmean[q];
      acc += w[9 + dy * 3 + dx] * pmax[q];
    }
  t1[i] = fmaxf(acc, 0.f);
}

__global__ void k_gate_conv2(const float* __restrict__ t1,
                             const float* __restrict__ w,
                             const float* __restrict__ bias,
                             float* __restrict__ logits) {
  int i = blockIdx.x * 256 + threadIdx.x;
  if (i >= B_ * L_) return;
  int b = i / L_, p = i % L_;
  int y = p / W_, x = p % W_;
  float acc = bias[0];
#pragma unroll
  for (int dy = 0; dy < 3; ++dy)
#pragma unroll
    for (int dx = 0; dx < 3; ++dx) {
      int yy = y + dy - 1, xx = x + dx - 1;
      if (yy < 0 || yy >= H_ || xx < 0 || xx >= W_) continue;
      acc += w[dy * 3 + dx] * t1[(size_t)b * L_ + yy * W_ + xx];
    }
  logits[i] = acc;
}

// ---------------------------------------------------------------------------
// Kernel 10: per-batch stable softmax over L=6400
// ---------------------------------------------------------------------------
__global__ __launch_bounds__(256) void k_softmax_L(
    const float* __restrict__ logits, float* __restrict__ g) {
  __shared__ float red[256];
  __shared__ float bmax, bsum;
  const int    b   = blockIdx.x;
  const int    tid = threadIdx.x;
  const float* x   = logits + (size_t)b * L_;

  float m = -FLT_MAX;
  for (int i = tid; i < L_; i += 256) m = fmaxf(m, x[i]);
  red[tid] = m;
  __syncthreads();
  for (int s = 128; s > 0; s >>= 1) {
    if (tid < s) red[tid] = fmaxf(red[tid], red[tid + s]);
    __syncthreads();
  }
  if (tid == 0) bmax = red[0];
  __syncthreads();

  float ssum = 0.f;
  for (int i = tid; i < L_; i += 256) ssum += __expf(x[i] - bmax);
  red[tid] = ssum;
  __syncthreads();
  for (int s = 128; s > 0; s >>= 1) {
    if (tid < s) red[tid] += red[tid + s];
    __syncthreads();
  }
  if (tid == 0) bsum = red[0];
  __syncthreads();

  float inv = 1.0f / bsum;
  for (int i = tid; i < L_; i += 256)
    g[(size_t)b * L_ + i] = __expf(x[i] - bmax) * inv;
}

// ---------------------------------------------------------------------------
// Kernel 11: out = f1*(1+g1) + f2*(1+g2)   (gates broadcast over channels)
// ---------------------------------------------------------------------------
__global__ void k_final(const float* __restrict__ f1,
                        const float* __restrict__ f2,
                        const float* __restrict__ g1,
                        const float* __restrict__ g2,
                        float* __restrict__ out) {
  size_t i = (size_t)blockIdx.x * 256 + threadIdx.x;
  if (i >= (size_t)B_ * C_ * L_) return;
  size_t b = i / ((size_t)C_ * L_);
  size_t l = i % L_;
  float  G1 = g1[b * L_ + l], G2 = g2[b * L_ + l];
  out[i] = f1[i] * (1.f + G1) + f2[i] * (1.f + G2);
}

// ---------------------------------------------------------------------------
// Host launcher
// ---------------------------------------------------------------------------
extern "C" void kernel_launch(void* const* d_in, const int* in_sizes, int n_in,
                              void* d_out, int out_size, void* d_ws,
                              size_t ws_size, hipStream_t stream) {
  const float* rgb        = (const float*)d_in[0];
  const float* freq       = (const float*)d_in[1];
  const float* la_qkv_w   = (const float*)d_in[2];
  const float* la_proj_w  = (const float*)d_in[3];
  const float* la_proj_b  = (const float*)d_in[4];
  const float* la_dconv_w = (const float*)d_in[5];
  const float* xa_qkv_w   = (const float*)d_in[6];
  const float* xa_temp    = (const float*)d_in[7];
  const float* xa_proj_w  = (const float*)d_in[8];
  const float* xa_proj_b  = (const float*)d_in[9];
  const float* c1s_w      = (const float*)d_in[10];
  const float* c1s_b      = (const float*)d_in[11];
  const float* c2s_w      = (const float*)d_in[12];
  const float* c2s_b      = (const float*)d_in[13];
  const float* avg1_w     = (const float*)d_in[14];
  const float* avg1_b     = (const float*)d_in[15];
  const float* max1_w     = (const float*)d_in[16];
  const float* max1_b     = (const float*)d_in[17];
  const float* avg2_w     = (const float*)d_in[18];
  const float* avg2_b     = (const float*)d_in[19];
  const float* max2_w     = (const float*)d_in[20];
  const float* max2_b     = (const float*)d_in[21];
  const float* avg11_w    = (const float*)d_in[22];
  const float* avg11_b    = (const float*)d_in[23];
  const float* max11_w    = (const float*)d_in[24];
  const float* max11_b    = (const float*)d_in[25];
  const float* avg22_w    = (const float*)d_in[26];
  const float* avg22_b    = (const float*)d_in[27];
  const float* max22_w    = (const float*)d_in[28];
  const float* max22_b    = (const float*)d_in[29];

  char* p    = (char*)d_ws;
  auto  take = [&](size_t bytes) -> void* {
    void* r = (void*)p;
    p += (bytes + 255) & ~(size_t)255;
    return r;
  };
  _Float16* q1n   = (_Float16*)take((size_t)B_ * L_ * 32 * 2);
  _Float16* v1    = (_Float16*)take((size_t)B_ * L_ * 32 * 2);
  _Float16* v2    = (_Float16*)take((size_t)B_ * L_ * 32 * 2);
  float*    f1    = (float*)take((size_t)B_ * C_ * L_ * 4);
  float*    f2    = (float*)take((size_t)B_ * C_ * L_ * 4);
  float*    attn1 = (float*)take((size_t)B_ * 128 * 4);
  float*    Sqk   = (float*)take((size_t)B_ * 128 * 4);
  float*    Sqq   = (float*)take((size_t)B_ * 32 * 4);
  float*    Skk   = (float*)take((size_t)B_ * 32 * 4);
  float*    attn2 = (float*)take((size_t)B_ * 128 * 4);
  float*    sum1  = (float*)take((size_t)B_ * 32 * 4);
  float*    max1  = (float*)take((size_t)B_ * 32 * 4);
  float*    sum2  = (float*)take((size_t)B_ * 32 * 4);
  float*    max2  = (float*)take((size_t)B_ * 32 * 4);
  float*    w1    = (float*)take((size_t)B_ * 1024 * 4);
  float*    w2    = (float*)take((size_t)B_ * 1024 * 4);
  float*    pm1   = (float*)take((size_t)B_ * L_ * 4);
  float*    px1   = (float*)take((size_t)B_ * L_ * 4);
  float*    pm2   = (float*)take((size_t)B_ * L_ * 4);
  float*    px2   = (float*)take((size_t)B_ * L_ * 4);
  float*    t1a   = (float*)take((size_t)B_ * L_ * 4);
  float*    t1b   = (float*)take((size_t)B_ * L_ * 4);
  float*    lg1   = (float*)take((size_t)B_ * L_ * 4);
  float*    lg2   = (float*)take((size_t)B_ * L_ * 4);
  float*    g1    = (float*)take((size_t)B_ * L_ * 4);
  float*    g2    = (float*)take((size_t)B_ * L_ * 4);

  k_fill<<<(B_ * 128 + 255) / 256, 256, 0, stream>>>(attn1, 0.f, B_ * 128);
  k_fill<<<(B_ * 128 + 255) / 256, 256, 0, stream>>>(Sqk, 0.f, B_ * 128);
  k_fill<<<(B_ * 32 + 255) / 256, 256, 0, stream>>>(Sqq, 0.f, B_ * 32);
  k_fill<<<(B_ * 32 + 255) / 256, 256, 0, stream>>>(Skk, 0.f, B_ * 32);
  k_fill<<<(B_ * 32 + 255) / 256, 256, 0, stream>>>(sum1, 0.f, B_ * 32);
  k_fill<<<(B_ * 32 + 255) / 256, 256, 0, stream>>>(sum2, 0.f, B_ * 32);
  k_fill<<<(B_ * 32 + 255) / 256, 256, 0, stream>>>(max1, -FLT_MAX, B_ * 32);
  k_fill<<<(B_ * 32 + 255) / 256, 256, 0, stream>>>(max2, -FLT_MAX, B_ * 32);

  dim3 gTile(L_ / 64, B_);
  k_qkv_la<<<gTile, 256, 0, stream>>>(rgb, la_qkv_w, q1n, v1, attn1);
  k_qkv_xca<<<gTile, 256, 0, stream>>>(freq, xa_qkv_w, v2, Sqk, Sqq, Skk);
  k_attn2<<<B_ * HEADS_, 16, 0, stream>>>(Sqk, Sqq, Skk, xa_temp, attn2);
  k_la_out<<<gTile, 256, 0, stream>>>(q1n, v1, attn1, la_dconv_w, la_proj_w,
                                      la_proj_b, f1, sum1, max1);
  k_xca_out<<<gTile, 256, 0, stream>>>(v2, attn2, xa_proj_w, xa_proj_b, f2,
                                       sum2, max2);
  k_chan_mlp<<<B_, 32, 0, stream>>>(
      sum1, max1, sum2, max2, avg1_w, avg1_b, max1_w, max1_b, avg11_w, avg11_b,
      max11_w, max11_b, avg2_w, avg2_b, max2_w, max2_b, avg22_w, avg22_b,
      max22_w, max22_b, w1, w2);

  dim3 gPool(L_ / 128, B_);
  k_gate_pool<<<gPool, 256, 0, stream>>>(w1, f1, pm1, px1);
  k_gate_pool<<<gPool, 256, 0, stream>>>(w2, f2, pm2, px2);

  int nBL = B_ * L_;
  k_gate_conv1<<<(nBL + 255) / 256, 256, 0, stream>>>(pm1, px1, c1s_w, c1s_b, t1a);
  k_gate_conv1<<<(nBL + 255) / 256, 256, 0, stream>>>(pm2, px2, c1s_w, c1s_b, t1b);
  k_gate_conv2<<<(nBL + 255) / 256, 256, 0, stream>>>(t1a, c2s_w, c2s_b, lg1);
  k_gate_conv2<<<(nBL + 255) / 256, 256, 0, stream>>>(t1b, c2s_w, c2s_b, lg2);
  k_softmax_L<<<B_, 256, 0, stream>>>(lg1, g1);
  k_softmax_L<<<B_, 256, 0, stream>>>(lg2, g2);

  size_t ntot = (size_t)B_ * C_ * L_;
  k_final<<<(unsigned)((ntot + 255) / 256), 256, 0, stream>>>(
      f1, f2, g1, g2, (float*)d_out);
}
